// OmegaPLMBatchedchkp_20426864460203
// MI455X (gfx1250) — compile-verified, matrix operand order checked by reading, not run on hardware
//
#include <hip/hip_runtime.h>
#include <hip/hip_bf16.h>
#include <math.h>

// ---------------- types ----------------
typedef __bf16 bf16;
typedef bf16 v16bf __attribute__((ext_vector_type(16)));
typedef bf16 v8bf  __attribute__((ext_vector_type(8)));
typedef float v8f  __attribute__((ext_vector_type(8)));

// ---------------- problem dims ----------------
#define NODE_D  1280
#define PROJ_D  2560
#define ATTN_D  256
#define GVA_N   5376   // 2*PROJ + ATTN
#define NREL_D  129
#define NL_D    4
#define B_D     2
#define L_D     1024
#define TOK_D   (B_D * L_D)   // 2048
#define ADALN_N 3840          // 3*NODE

// ---------------- WMMA GEMM tiling ----------------
// Block tile 256(M) x 128(N), 8 wave32 waves in a 4x2 grid of 64x64 wave tiles.
// Per wave per K-step: 16 ds_load_b128 feed 16 v_wmma (1.0 wmma/load vs 0.67
// for 32x64) -> the matrix pipe, not LDS bandwidth, is the limiter.
#define BM 256
#define BN 128
#define BK 32
#define LDSK 40              // padded LDS row stride (80B): conflict-free b128 frag loads
#define AT_E (BM * LDSK)     // elems per A LDS buffer
#define BT_E (BN * LDSK)     // elems per B LDS buffer

// Load one 16x32 bf16 fragment (A layout per CDNA5 ISA 7.12.2) from an LDS tile.
// lane = (m within 16) + 16*g ; holds K in [8g,8g+8) U [16+8g, 16+8g+8).
__device__ __forceinline__ void frag_from_lds(const bf16* base, int row, int g, v16bf& f) {
  const v8bf lo = *(const v8bf*)(base + row * LDSK + 8 * g);
  const v8bf hi = *(const v8bf*)(base + row * LDSK + 16 + 8 * g);
#pragma unroll
  for (int i = 0; i < 8; ++i) { f[i] = lo[i]; f[i + 8] = hi[i]; }
}

__device__ __forceinline__ void compute_tile(
    const bf16* Ac, const bf16* Bc, int wm, int wn, int g, int l16, v8f acc[4][4])
{
  v16bf af[4], bfr[4];
#pragma unroll
  for (int r = 0; r < 4; ++r) frag_from_lds(Ac, wm + r * 16 + l16, g, af[r]);
#pragma unroll
  for (int c = 0; c < 4; ++c) frag_from_lds(Bc, wn + c * 16 + l16, g, bfr[c]);
#pragma unroll
  for (int r = 0; r < 4; ++r)
#pragma unroll
    for (int c = 0; c < 4; ++c)
      acc[r][c] = __builtin_amdgcn_wmma_f32_16x16x32_bf16(
          false, af[r], false, bfr[c], (short)0, acc[r][c], false, false);
}

// C = A(MxK, row-major, bf16) * Bt(NxK, row-major, bf16)^T, fp32 accum.
// Double-buffered LDS, register-staged global prefetch: steady-state body is
// {global_load tile k -> regs; 16x WMMA on tile k-1; ds_store regs; 1 barrier}.
__device__ __forceinline__ void gemm_loop(
    const bf16* __restrict__ A, int lda, int m0,
    const bf16* __restrict__ Bt, int ldb, int n0,
    int K, bf16* As, bf16* Bs, v8f acc[4][4])
{
  const int tid  = threadIdx.x;
  const int lane = tid & 31;
  const int wv   = tid >> 5;
  const int wm   = (wv >> 1) * 64;   // 4 waves along M (256)
  const int wn   = (wv & 1) * 64;    // 2 waves along N (128)
  const int g    = lane >> 4;
  const int l16  = lane & 15;

#pragma unroll
  for (int r = 0; r < 4; ++r)
#pragma unroll
    for (int c = 0; c < 4; ++c)
#pragma unroll
      for (int v = 0; v < 8; ++v) acc[r][c][v] = 0.0f;

  const int rowA = tid >> 2;       // 0..63 (row-groups of 64)
  const int qA   = (tid & 3) * 8;  // bf16 element offset within K-chunk
  const int lo   = rowA * LDSK + qA;

  const bf16* gA = A  + (size_t)(m0 + rowA) * lda + qA;
  const bf16* gB = Bt + (size_t)(n0 + rowA) * ldb + qA;

  const int nk = K / BK;
  uint4 ra[4], rb[2];

  // ---- prolog: stage tile 0 into buffer 0 ----
#pragma unroll
  for (int j = 0; j < 4; ++j) ra[j] = *(const uint4*)(gA + (size_t)(j * 64) * lda);
#pragma unroll
  for (int j = 0; j < 2; ++j) rb[j] = *(const uint4*)(gB + (size_t)(j * 64) * ldb);
#pragma unroll
  for (int j = 0; j < 4; ++j) *(uint4*)(As + lo + j * 64 * LDSK) = ra[j];
#pragma unroll
  for (int j = 0; j < 2; ++j) *(uint4*)(Bs + lo + j * 64 * LDSK) = rb[j];
  __syncthreads();

  // ---- steady state ----
  for (int kb = 1; kb < nk; ++kb) {
    // issue global loads for tile kb early (latency hides under WMMA below)
#pragma unroll
    for (int j = 0; j < 4; ++j) ra[j] = *(const uint4*)(gA + (size_t)(j * 64) * lda + kb * BK);
#pragma unroll
    for (int j = 0; j < 2; ++j) rb[j] = *(const uint4*)(gB + (size_t)(j * 64) * ldb + kb * BK);

    compute_tile(As + ((kb - 1) & 1) * AT_E, Bs + ((kb - 1) & 1) * BT_E,
                 wm, wn, g, l16, acc);

    bf16* An = As + (kb & 1) * AT_E;
    bf16* Bn = Bs + (kb & 1) * BT_E;
#pragma unroll
    for (int j = 0; j < 4; ++j) *(uint4*)(An + lo + j * 64 * LDSK) = ra[j];
#pragma unroll
    for (int j = 0; j < 2; ++j) *(uint4*)(Bn + lo + j * 64 * LDSK) = rb[j];
    __syncthreads();
  }

  // ---- epilog: consume last tile ----
  compute_tile(As + ((nk - 1) & 1) * AT_E, Bs + ((nk - 1) & 1) * BT_E,
               wm, wn, g, l16, acc);
}

// epilogue coordinates for frag (r,c), element v:
//   m_local = wm + r*16 + g*8 + v ;  n_local = wn + c*16 + l16
#define EPILOGUE_SETUP()                       \
  const int lane = threadIdx.x & 31;           \
  const int wv   = threadIdx.x >> 5;           \
  const int wm   = (wv >> 1) * 64;             \
  const int wn   = (wv & 1) * 64;              \
  const int g    = lane >> 4;                  \
  const int l16  = lane & 15;

__device__ __forceinline__ float silu_f(float x) { return x / (1.0f + __expf(-x)); }

// ---------------- GEMM 1: h = silu(x @ w_gva + b_gva), fp32 out ----------------
__global__ __launch_bounds__(256) void gemm_gva_kernel(
    const bf16* __restrict__ xbf, const bf16* __restrict__ wgva_t,
    const float* __restrict__ b_gva, float* __restrict__ h, int l)
{
  __shared__ bf16 As[2 * AT_E];
  __shared__ bf16 Bs[2 * BT_E];
  v8f acc[4][4];
  const int m0 = blockIdx.y * BM, n0 = blockIdx.x * BN;
  gemm_loop(xbf, NODE_D, m0, wgva_t + (size_t)l * GVA_N * NODE_D, NODE_D, n0,
            NODE_D, As, Bs, acc);
  EPILOGUE_SETUP();
#pragma unroll
  for (int r = 0; r < 4; ++r)
#pragma unroll
    for (int c = 0; c < 4; ++c)
#pragma unroll
      for (int v = 0; v < 8; ++v) {
        const int m = m0 + wm + r * 16 + g * 8 + v;
        const int n = n0 + wn + c * 16 + l16;
        h[(size_t)m * GVA_N + n] = silu_f(acc[r][c][v] + b_gva[l * GVA_N + n]);
      }
}

// ---------------- GEMM 2: logits = qs @ k^T + bias + relpos ----------------
__global__ __launch_bounds__(256) void gemm_logits_kernel(
    const bf16* __restrict__ qs, const bf16* __restrict__ kb,
    const float* __restrict__ bias, const float* __restrict__ relpos,
    float* __restrict__ logits, int l)
{
  __shared__ bf16 As[2 * AT_E];
  __shared__ bf16 Bs[2 * BT_E];
  v8f acc[4][4];
  const int b  = blockIdx.z;
  const int m0 = blockIdx.y * BM, n0 = blockIdx.x * BN;
  gemm_loop(qs + (size_t)b * L_D * ATTN_D, ATTN_D, m0,
            kb + (size_t)b * L_D * ATTN_D, ATTN_D, n0, ATTN_D, As, Bs, acc);
  EPILOGUE_SETUP();
#pragma unroll
  for (int r = 0; r < 4; ++r)
#pragma unroll
    for (int c = 0; c < 4; ++c)
#pragma unroll
      for (int v = 0; v < 8; ++v) {
        const int i = m0 + wm + r * 16 + g * 8 + v;
        const int j = n0 + wn + c * 16 + l16;
        int d = i - j; d = (d < -64) ? -64 : (d > 64 ? 64 : d); d += 64;
        logits[((size_t)b * L_D + i) * L_D + j] =
            acc[r][c][v] + bias[b * L_D + j] + relpos[l * NREL_D + d];
      }
}

// ---------------- GEMM 3: a = (attn @ values) * gates -> bf16 ----------------
__global__ __launch_bounds__(256) void gemm_av_kernel(
    const bf16* __restrict__ probbf, const bf16* __restrict__ vt,
    const float* __restrict__ h, bf16* __restrict__ abf)
{
  __shared__ bf16 As[2 * AT_E];
  __shared__ bf16 Bs[2 * BT_E];
  v8f acc[4][4];
  const int b  = blockIdx.z;
  const int m0 = blockIdx.y * BM, n0 = blockIdx.x * BN;
  gemm_loop(probbf + (size_t)b * L_D * L_D, L_D, m0,
            vt + (size_t)b * PROJ_D * L_D, L_D, n0, L_D, As, Bs, acc);
  EPILOGUE_SETUP();
#pragma unroll
  for (int r = 0; r < 4; ++r)
#pragma unroll
    for (int c = 0; c < 4; ++c)
#pragma unroll
      for (int v = 0; v < 8; ++v) {
        const int tok = b * L_D + m0 + wm + r * 16 + g * 8 + v;
        const int n   = n0 + wn + c * 16 + l16;   // 0..2559: gates region of h
        const float gate = h[(size_t)tok * GVA_N + n];
        abf[(size_t)tok * PROJ_D + n] = (bf16)(acc[r][c][v] * gate);
      }
}

// ---------------- GEMM 4: node = (a @ w_out + b_out)*(1+alpha) + shortcut ----
__global__ __launch_bounds__(256) void gemm_out_kernel(
    const bf16* __restrict__ abf, const bf16* __restrict__ wout_t,
    const float* __restrict__ b_out, const float* __restrict__ gba,
    const float* __restrict__ shortcut, float* __restrict__ outp, int l)
{
  __shared__ bf16 As[2 * AT_E];
  __shared__ bf16 Bs[2 * BT_E];
  v8f acc[4][4];
  const int m0 = blockIdx.y * BM, n0 = blockIdx.x * BN;
  gemm_loop(abf, PROJ_D, m0, wout_t + (size_t)l * NODE_D * PROJ_D, PROJ_D, n0,
            PROJ_D, As, Bs, acc);
  EPILOGUE_SETUP();
#pragma unroll
  for (int r = 0; r < 4; ++r)
#pragma unroll
    for (int c = 0; c < 4; ++c)
#pragma unroll
      for (int v = 0; v < 8; ++v) {
        const int tok = m0 + wm + r * 16 + g * 8 + v;
        const int b   = tok >> 10;
        const int n   = n0 + wn + c * 16 + l16;
        const float alpha = gba[b * ADALN_N + 2 * NODE_D + n];
        float o = acc[r][c][v] + b_out[l * NODE_D + n];
        outp[(size_t)tok * NODE_D + n] =
            o * (1.0f + alpha) + shortcut[(size_t)tok * NODE_D + n];
      }
}

// ---------------- adaLN: gba = silu(cond) @ w_adaln + b_adaln ----------------
__global__ __launch_bounds__(256) void adaln_kernel(
    const float* __restrict__ cond, const float* __restrict__ w,
    const float* __restrict__ bia, float* __restrict__ gba, int l)
{
  const int c = blockIdx.x * 256 + threadIdx.x;  // 0..3839
  const int b = blockIdx.y;
  const float* wl = w + (size_t)l * NODE_D * ADALN_N;
  float s = bia[l * ADALN_N + c];
  for (int k = 0; k < NODE_D; ++k)
    s += silu_f(cond[b * NODE_D + k]) * wl[(size_t)k * ADALN_N + c];
  gba[b * ADALN_N + c] = s;
}

// ---------------- LN + modulate -> bf16 x ----------------
__global__ __launch_bounds__(256) void ln_kernel(
    const float* __restrict__ node, const float* __restrict__ gba,
    bf16* __restrict__ xbf)
{
  __shared__ float r1[256];
  __shared__ float r2[256];
  const int tok = blockIdx.x;
  const int b   = tok >> 10;
  const float* row = node + (size_t)tok * NODE_D;
  float s = 0.f, s2 = 0.f;
  for (int c = threadIdx.x; c < NODE_D; c += 256) { float v = row[c]; s += v; s2 += v * v; }
  r1[threadIdx.x] = s; r2[threadIdx.x] = s2; __syncthreads();
  for (int o = 128; o > 0; o >>= 1) {
    if (threadIdx.x < o) { r1[threadIdx.x] += r1[threadIdx.x + o]; r2[threadIdx.x] += r2[threadIdx.x + o]; }
    __syncthreads();
  }
  const float mean = r1[0] * (1.0f / NODE_D);
  const float var  = r2[0] * (1.0f / NODE_D) - mean * mean;
  const float inv  = rsqrtf(var + 1e-5f);
  for (int c = threadIdx.x; c < NODE_D; c += 256) {
    const float ga = gba[b * ADALN_N + c];
    const float be = gba[b * ADALN_N + NODE_D + c];
    const float xn = (row[c] - mean) * inv;
    xbf[(size_t)tok * NODE_D + c] = (bf16)(xn * (1.0f + ga) + be);
  }
}

// ---------------- mhs scale + RoPE + qk_scaling -> bf16 q,k ----------------
__global__ __launch_bounds__(128) void rope_kernel(
    const float* __restrict__ h, const float* __restrict__ qk_scaling,
    const float* __restrict__ mhs_w, const float* __restrict__ mhs_b,
    bf16* __restrict__ qs, bf16* __restrict__ kb, int l)
{
  const int tok = blockIdx.x;
  const int i   = tok & (L_D - 1);
  const int t   = threadIdx.x;  // 0..127
  const float* base = h + (size_t)tok * GVA_N + 2 * PROJ_D;
  const float b0 = base[t], b1 = base[t + 128];
  const float* mw = mhs_w + l * 2 * ATTN_D;
  const float* mb = mhs_b + l * 2 * ATTN_D;
  const float q0 = b0 * mw[t] + mb[t];
  const float q1 = b1 * mw[t + 128] + mb[t + 128];
  const float k0 = b0 * mw[256 + t] + mb[256 + t];
  const float k1 = b1 * mw[256 + t + 128] + mb[256 + t + 128];
  const float fr = __expf(-((float)(2 * t) * (1.0f / 256.0f)) * 9.210340371976184f);
  const float ang = (float)i * fr;
  const float cs = cosf(ang), sn = sinf(ang);
  const float sc = qk_scaling[tok];
  qs[(size_t)tok * ATTN_D + t]       = (bf16)((q0 * cs - q1 * sn) * sc);
  qs[(size_t)tok * ATTN_D + t + 128] = (bf16)((q1 * cs + q0 * sn) * sc);
  kb[(size_t)tok * ATTN_D + t]       = (bf16)(k0 * cs - k1 * sn);
  kb[(size_t)tok * ATTN_D + t + 128] = (bf16)(k1 * cs + k0 * sn);
}

// ---------------- row softmax -> bf16 probs (+fp32 on last layer) ----------------
__global__ __launch_bounds__(256) void softmax_kernel(
    const float* __restrict__ logits, bf16* __restrict__ pbf,
    float* __restrict__ pf32, int storeF32)
{
  __shared__ float red[256];
  const int row = blockIdx.x;  // b*1024 + i
  const float* lr = logits + (size_t)row * L_D;
  float v[4]; float mx = -3.4e38f;
#pragma unroll
  for (int k = 0; k < 4; ++k) { v[k] = lr[threadIdx.x + k * 256]; mx = fmaxf(mx, v[k]); }
  red[threadIdx.x] = mx; __syncthreads();
  for (int o = 128; o > 0; o >>= 1) {
    if (threadIdx.x < o) red[threadIdx.x] = fmaxf(red[threadIdx.x], red[threadIdx.x + o]);
    __syncthreads();
  }
  mx = red[0]; __syncthreads();
  float s = 0.f;
#pragma unroll
  for (int k = 0; k < 4; ++k) { v[k] = __expf(v[k] - mx); s += v[k]; }
  red[threadIdx.x] = s; __syncthreads();
  for (int o = 128; o > 0; o >>= 1) {
    if (threadIdx.x < o) red[threadIdx.x] += red[threadIdx.x + o];
    __syncthreads();
  }
  const float inv = 1.0f / red[0];
#pragma unroll
  for (int k = 0; k < 4; ++k) {
    const float p = v[k] * inv;
    pbf[(size_t)row * L_D + threadIdx.x + k * 256] = (bf16)p;
    if (storeF32) pf32[(size_t)row * L_D + threadIdx.x + k * 256] = p;
  }
}

// ---------------- fp32 MxN -> bf16 NxM transpose-convert ----------------
__global__ void transpose_cvt_kernel(
    const float* __restrict__ in, bf16* __restrict__ out,
    int ldin, int ldout, size_t inBatchStride, size_t outBatchStride)
{
  __shared__ float tile[32][33];
  const float* ip = in + blockIdx.z * inBatchStride;
  bf16* op = out + blockIdx.z * outBatchStride;
  const int c0 = blockIdx.x * 32, r0 = blockIdx.y * 32;
#pragma unroll
  for (int i = 0; i < 4; ++i)
    tile[threadIdx.y + i * 8][threadIdx.x] =
        ip[(size_t)(r0 + threadIdx.y + i * 8) * ldin + c0 + threadIdx.x];
  __syncthreads();
#pragma unroll
  for (int i = 0; i < 4; ++i)
    op[(size_t)(c0 + threadIdx.y + i * 8) * ldout + r0 + threadIdx.x] =
        (bf16)tile[threadIdx.x][threadIdx.y + i * 8];
}

__global__ void copy_f32_kernel(const float* __restrict__ in, float* __restrict__ out, int n) {
  const int i = blockIdx.x * 256 + threadIdx.x;
  if (i < n) out[i] = in[i];
}

__global__ void edge_kernel(const float* __restrict__ pf32, float* __restrict__ edge) {
  const int i = blockIdx.x * 256 + threadIdx.x;
  edge[i] = pf32[i] + pf32[(size_t)L_D * L_D + i];
}

// ---------------- launcher ----------------
extern "C" void kernel_launch(void* const* d_in, const int* in_sizes, int n_in,
                              void* d_out, int out_size, void* d_ws, size_t ws_size,
                              hipStream_t stream) {
  const float* node_in    = (const float*)d_in[0];
  const float* qk_scaling = (const float*)d_in[1];
  const float* bias       = (const float*)d_in[2];
  const float* cond       = (const float*)d_in[3];
  const float* w_gva      = (const float*)d_in[4];
  const float* b_gva      = (const float*)d_in[5];
  const float* mhs_w      = (const float*)d_in[6];
  const float* mhs_b      = (const float*)d_in[7];
  const float* relpos     = (const float*)d_in[8];
  const float* w_out      = (const float*)d_in[9];
  const float* b_out      = (const float*)d_in[10];
  const float* w_adaln    = (const float*)d_in[11];
  const float* b_adaln    = (const float*)d_in[12];

  char* wsp = (char*)d_ws;
  auto alloc = [&](size_t bytes) -> char* {
    char* p = wsp;
    wsp += (bytes + 255) & ~(size_t)255;
    return p;
  };
  bf16*  wgva_t  = (bf16*)alloc((size_t)NL_D * GVA_N * NODE_D * sizeof(bf16));
  bf16*  wout_t  = (bf16*)alloc((size_t)NL_D * NODE_D * PROJ_D * sizeof(bf16));
  float* nodebuf = (float*)alloc((size_t)TOK_D * NODE_D * sizeof(float));
  bf16*  xbf     = (bf16*)alloc((size_t)TOK_D * NODE_D * sizeof(bf16));
  float* gba     = (float*)alloc((size_t)B_D * ADALN_N * sizeof(float));
  float* hf32    = (float*)alloc((size_t)TOK_D * GVA_N * sizeof(float));
  bf16*  qsbf    = (bf16*)alloc((size_t)TOK_D * ATTN_D * sizeof(bf16));
  bf16*  kbf     = (bf16*)alloc((size_t)TOK_D * ATTN_D * sizeof(bf16));
  float* logits  = (float*)alloc((size_t)B_D * L_D * L_D * sizeof(float));
  bf16*  probbf  = (bf16*)alloc((size_t)B_D * L_D * L_D * sizeof(bf16));
  float* probf32 = (float*)alloc((size_t)B_D * L_D * L_D * sizeof(float));
  bf16*  vt      = (bf16*)alloc((size_t)B_D * PROJ_D * L_D * sizeof(bf16));
  bf16*  abf     = (bf16*)alloc((size_t)TOK_D * PROJ_D * sizeof(bf16));

  float* out_node = (float*)d_out;                           // (2,1024,1280)
  float* out_edge = (float*)d_out + (size_t)TOK_D * NODE_D;  // (1024,1024)

  // One-time: weight transpose+convert to bf16 (Nt x K) and node copy.
  transpose_cvt_kernel<<<dim3(GVA_N / 32, NODE_D / 32, NL_D), dim3(32, 8), 0, stream>>>(
      w_gva, wgva_t, GVA_N, NODE_D, (size_t)NODE_D * GVA_N, (size_t)NODE_D * GVA_N);
  transpose_cvt_kernel<<<dim3(NODE_D / 32, PROJ_D / 32, NL_D), dim3(32, 8), 0, stream>>>(
      w_out, wout_t, NODE_D, PROJ_D, (size_t)PROJ_D * NODE_D, (size_t)PROJ_D * NODE_D);
  copy_f32_kernel<<<(TOK_D * NODE_D + 255) / 256, 256, 0, stream>>>(
      node_in, nodebuf, TOK_D * NODE_D);

  for (int l = 0; l < NL_D; ++l) {
    adaln_kernel<<<dim3(ADALN_N / 256, B_D), 256, 0, stream>>>(cond, w_adaln, b_adaln, gba, l);
    ln_kernel<<<TOK_D, 256, 0, stream>>>(nodebuf, gba, xbf);
    gemm_gva_kernel<<<dim3(GVA_N / BN, TOK_D / BM), 256, 0, stream>>>(
        xbf, wgva_t, b_gva, hf32, l);
    rope_kernel<<<TOK_D, 128, 0, stream>>>(hf32, qk_scaling, mhs_w, mhs_b, qsbf, kbf, l);
    gemm_logits_kernel<<<dim3(L_D / BN, L_D / BM, B_D), 256, 0, stream>>>(
        qsbf, kbf, bias, relpos, logits, l);
    softmax_kernel<<<B_D * L_D, 256, 0, stream>>>(logits, probbf, probf32, (l == NL_D - 1) ? 1 : 0);
    // values (1024x2560 fp32 slice of h) -> vt (2560x1024 bf16), per batch
    transpose_cvt_kernel<<<dim3(PROJ_D / 32, L_D / 32, B_D), dim3(32, 8), 0, stream>>>(
        hf32 + PROJ_D, vt, GVA_N, L_D, (size_t)L_D * GVA_N, (size_t)PROJ_D * L_D);
    gemm_av_kernel<<<dim3(PROJ_D / BN, L_D / BM, B_D), 256, 0, stream>>>(probbf, vt, hf32, abf);
    float* outp = (l == NL_D - 1) ? out_node : nodebuf;
    gemm_out_kernel<<<dim3(NODE_D / BN, TOK_D / BM), 256, 0, stream>>>(
        abf, wout_t, b_out, gba, nodebuf, outp, l);
  }
  edge_kernel<<<(L_D * L_D) / 256, 256, 0, stream>>>(probf32, out_edge);
}
